// SharedSpecialistMoEFFN_27384711479465
// MI455X (gfx1250) — matrix operand order repaired and checked by reading, not compiled, not run on hardware
//
#include <hip/hip_runtime.h>
#include <hip/hip_bf16.h>

#define U_TOK 8192
#define DM    1024
#define DFF   4096
#define NEXP  8

typedef __attribute__((ext_vector_type(16))) __bf16 v16bf;
typedef __attribute__((ext_vector_type(8)))  float  v8f;
typedef __attribute__((ext_vector_type(4)))  __bf16 bf16x4;

__device__ __forceinline__ int imin_(int a, int b){ return a < b ? a : b; }

__device__ __forceinline__ float gelu_tanh(float x){
  float x3 = x * x * x;
  return 0.5f * x * (1.0f + tanhf(0.7978845608028654f * (x + 0.044715f * x3)));
}

// Load one 16-element bf16 fragment as two 16B chunks (gap in elements).
__device__ __forceinline__ v16bf frag_ld(const __bf16* p, int gap){
  union { uint4 u[2]; v16bf v; } r;
  r.u[0] = *(const uint4*)p;
  r.u[1] = *(const uint4*)(p + gap);
  return r.v;
}

__device__ __forceinline__ v8f wmma_bf16(v16bf a, v16bf b, v8f c){
  return __builtin_amdgcn_wmma_f32_16x16x32_bf16(false, a, false, b, (short)0, c, false, false);
}

// ---------------- fp32 -> bf16 convert ----------------
__global__ void k_cvt(const float4* __restrict__ src, bf16x4* __restrict__ dst, int n4){
  int stride = gridDim.x * blockDim.x;
  for (int i = blockIdx.x * blockDim.x + threadIdx.x; i < n4; i += stride){
    float4 f = src[i];
    bf16x4 b;
    b[0] = (__bf16)f.x; b[1] = (__bf16)f.y; b[2] = (__bf16)f.z; b[3] = (__bf16)f.w;
    dst[i] = b;
  }
}

// ---------------- router: logits -> softmax -> top2 -> renorm ----------------
__global__ __launch_bounds__(256) void k_router(const float* __restrict__ x,
                                                const float* __restrict__ rw,
                                                const float* __restrict__ rb,
                                                int* __restrict__ tk_idx,
                                                float* __restrict__ tk_w){
  int lane = threadIdx.x & 31;
  int wid  = threadIdx.x >> 5;
  int t = blockIdx.x * 8 + wid;
  if (t >= U_TOK) return;
  float acc[NEXP];
#pragma unroll
  for (int e = 0; e < NEXP; ++e) acc[e] = 0.0f;
  const float* xp = x + (size_t)t * DM;
  for (int d = lane; d < DM; d += 32){
    float xv = xp[d];
#pragma unroll
    for (int e = 0; e < NEXP; ++e) acc[e] += xv * rw[e * DM + d];
  }
#pragma unroll
  for (int e = 0; e < NEXP; ++e){
#pragma unroll
    for (int o = 16; o > 0; o >>= 1) acc[e] += __shfl_xor(acc[e], o, 32);
  }
  if (lane == 0){
    float lg[NEXP];
    float m = -3.4e38f;
#pragma unroll
    for (int e = 0; e < NEXP; ++e){ lg[e] = acc[e] + rb[e]; m = fmaxf(m, lg[e]); }
    float s = 0.0f;
#pragma unroll
    for (int e = 0; e < NEXP; ++e){ lg[e] = expf(lg[e] - m); s += lg[e]; }
    float inv = 1.0f / s;
#pragma unroll
    for (int e = 0; e < NEXP; ++e) lg[e] *= inv;
    int i0 = 0; float p0 = lg[0];
#pragma unroll
    for (int e = 1; e < NEXP; ++e) if (lg[e] > p0){ p0 = lg[e]; i0 = e; }
    int i1 = -1; float p1 = -1.0f;
#pragma unroll
    for (int e = 0; e < NEXP; ++e) if (e != i0 && lg[e] > p1){ p1 = lg[e]; i1 = e; }
    float rs = 1.0f / (p0 + p1);
    tk_idx[t * 2 + 0] = i0; tk_w[t * 2 + 0] = p0 * rs;
    tk_idx[t * 2 + 1] = i1; tk_w[t * 2 + 1] = p1 * rs;
  }
}

// ---------------- build per-expert token lists ----------------
__global__ __launch_bounds__(256) void k_build(const int* __restrict__ tk_idx,
                                               const float* __restrict__ tk_w,
                                               int* __restrict__ etok,
                                               float* __restrict__ egate,
                                               int* __restrict__ eoff){
  __shared__ int cnt[NEXP], cur[NEXP], off[NEXP + 1];
  int tid = threadIdx.x;
  if (tid < NEXP) cnt[tid] = 0;
  __syncthreads();
  for (int i = tid; i < U_TOK * 2; i += blockDim.x) atomicAdd(&cnt[tk_idx[i]], 1);
  __syncthreads();
  if (tid == 0){
    off[0] = 0;
    for (int e = 0; e < NEXP; ++e){ off[e + 1] = off[e] + cnt[e]; cur[e] = off[e]; }
    for (int e = 0; e <= NEXP; ++e) eoff[e] = off[e];
  }
  __syncthreads();
  for (int i = tid; i < U_TOK * 2; i += blockDim.x){
    int e = tk_idx[i];
    int p = atomicAdd(&cur[e], 1);
    etok[p]  = i >> 1;
    egate[p] = tk_w[i];
  }
}

// ==================== GEMM core geometry ====================
// Block tile: 128(M) x 128(N), K-stage 32, double-buffered LDS.
// 8 waves: mw = wid&3 (M quadrant of 32), nw = wid>>2 (N half of 64).
// Each wave: 2 A-frags x 4 B-frags -> 8 WMMAs per stage, 1 barrier per stage.

// ---------------- GEMM1: h = gelu(gather(xb) @ w1^T + b1), bf16 out ----------------
__global__ __launch_bounds__(256) void k_ffn1(const __bf16* __restrict__ xb,
                                              const __bf16* __restrict__ w1,
                                              const float*  __restrict__ b1,
                                              __bf16* __restrict__ h,
                                              const int* __restrict__ etok,
                                              const int* __restrict__ eoff,
                                              int expert){
  __shared__ __bf16 As[2][128][40];
  __shared__ __bf16 Bs[2][128][40];
  __shared__ int stok[128];

  int mb = blockIdx.x, nb = blockIdx.y;
  int base = 0, cnt = U_TOK;
  if (expert >= 0){ base = eoff[expert]; cnt = eoff[expert + 1] - base; }
  if (mb * 128 >= cnt) return;

  int tid = threadIdx.x;
  if (tid < 128){
    int gm = imin_(mb * 128 + tid, cnt - 1);
    stok[tid] = (expert >= 0) ? etok[base + gm] : gm;
  }
  __syncthreads();

  int lane = tid & 31, wid = tid >> 5;
  int mw = wid & 3, nw = wid >> 2;
  int aoff  = (lane & 16) ? 16 : 0;  // A fragment K-half offset
  int koffB = (lane & 16) ? 16 : 0;  // B fragment K-half offset
  int arow  = lane & 15;
  int ncol  = lane & 15;

  v8f acc[2][4];
#pragma unroll
  for (int i = 0; i < 2; ++i)
#pragma unroll
    for (int j = 0; j < 4; ++j) acc[i][j] = (v8f){0,0,0,0,0,0,0,0};

  // per-thread staging indices: 512 uint4 per operand tile, 2 per thread
  const int r0 = (tid * 2)     >> 2, c0 = ((tid * 2)     & 3) << 3;
  const int r1 = (tid * 2 + 1) >> 2, c1 = ((tid * 2 + 1) & 3) << 3;
  const __bf16* ap0 = xb + (size_t)stok[r0] * DM + c0;
  const __bf16* ap1 = xb + (size_t)stok[r1] * DM + c1;
  const __bf16* bp0 = w1 + (size_t)(nb * 128 + r0) * DM + c0;
  const __bf16* bp1 = w1 + (size_t)(nb * 128 + r1) * DM + c1;

  // prologue: stage k=0 into buffer 0
  {
    uint4 a0v = *(const uint4*)ap0;
    uint4 a1v = *(const uint4*)ap1;
    uint4 b0v = *(const uint4*)bp0;
    uint4 b1v = *(const uint4*)bp1;
    *(uint4*)(&As[0][r0][c0]) = a0v;
    *(uint4*)(&As[0][r1][c1]) = a1v;
    *(uint4*)(&Bs[0][r0][c0]) = b0v;
    *(uint4*)(&Bs[0][r1][c1]) = b1v;
  }
  __syncthreads();

  const int NK = DM / 32;
  for (int ki = 0; ki < NK; ++ki){
    int cur = ki & 1, nxt = cur ^ 1;
    int kn = imin_((ki + 1) * 32, DM - 32);   // clamped: branch-free last stage
    // stage next K-slice into the other buffer (safe: last read of it was
    // before the barrier at the end of the previous iteration)
    uint4 a0v = *(const uint4*)(ap0 + kn);
    uint4 a1v = *(const uint4*)(ap1 + kn);
    uint4 b0v = *(const uint4*)(bp0 + kn);
    uint4 b1v = *(const uint4*)(bp1 + kn);
    *(uint4*)(&As[nxt][r0][c0]) = a0v;
    *(uint4*)(&As[nxt][r1][c1]) = a1v;
    *(uint4*)(&Bs[nxt][r0][c0]) = b0v;
    *(uint4*)(&Bs[nxt][r1][c1]) = b1v;

    v16bf a0 = frag_ld(&As[cur][mw * 32 + arow][aoff], 8);
    v16bf a1 = frag_ld(&As[cur][mw * 32 + 16 + arow][aoff], 8);
    v16bf bb0 = frag_ld(&Bs[cur][nw * 64 +  0 + ncol][koffB], 8);
    v16bf bb1 = frag_ld(&Bs[cur][nw * 64 + 16 + ncol][koffB], 8);
    v16bf bb2 = frag_ld(&Bs[cur][nw * 64 + 32 + ncol][koffB], 8);
    v16bf bb3 = frag_ld(&Bs[cur][nw * 64 + 48 + ncol][koffB], 8);
    acc[0][0] = wmma_bf16(a0, bb0, acc[0][0]);
    acc[0][1] = wmma_bf16(a0, bb1, acc[0][1]);
    acc[0][2] = wmma_bf16(a0, bb2, acc[0][2]);
    acc[0][3] = wmma_bf16(a0, bb3, acc[0][3]);
    acc[1][0] = wmma_bf16(a1, bb0, acc[1][0]);
    acc[1][1] = wmma_bf16(a1, bb1, acc[1][1]);
    acc[1][2] = wmma_bf16(a1, bb2, acc[1][2]);
    acc[1][3] = wmma_bf16(a1, bb3, acc[1][3]);
    __syncthreads();
  }

#pragma unroll
  for (int mi = 0; mi < 2; ++mi)
#pragma unroll
    for (int ni = 0; ni < 4; ++ni){
      int col = nb * 128 + nw * 64 + ni * 16 + (lane & 15);
      float bias = b1[col];
#pragma unroll
      for (int v = 0; v < 8; ++v){
        int row = mw * 32 + mi * 16 + v + ((lane & 16) ? 8 : 0);
        int gm = mb * 128 + row;
        if (gm < cnt){
          float val = acc[mi][ni][v] + bias;
          h[(size_t)gm * DFF + col] = (__bf16)gelu_tanh(val);
        }
      }
    }
}

// ---------------- GEMM2: out[token] (+)= gate*(h @ w2^T + b2) ----------------
__global__ __launch_bounds__(256) void k_ffn2(const __bf16* __restrict__ h,
                                              const __bf16* __restrict__ w2,
                                              const float*  __restrict__ b2,
                                              float* __restrict__ out,
                                              const int* __restrict__ etok,
                                              const float* __restrict__ egate,
                                              const int* __restrict__ eoff,
                                              int expert){
  __shared__ __bf16 As[2][128][40];
  __shared__ __bf16 Bs[2][128][40];
  __shared__ int   stok[128];
  __shared__ float sgate[128];

  int mb = blockIdx.x, nb = blockIdx.y;
  int base = 0, cnt = U_TOK;
  if (expert >= 0){ base = eoff[expert]; cnt = eoff[expert + 1] - base; }
  if (mb * 128 >= cnt) return;

  int tid = threadIdx.x;
  if (tid < 128){
    int gm = imin_(mb * 128 + tid, cnt - 1);
    if (expert >= 0){ stok[tid] = etok[base + gm]; sgate[tid] = egate[base + gm]; }
    else            { stok[tid] = gm;              sgate[tid] = 1.0f; }
  }
  __syncthreads();

  int lane = tid & 31, wid = tid >> 5;
  int mw = wid & 3, nw = wid >> 2;
  int aoff  = (lane & 16) ? 16 : 0;
  int koffB = (lane & 16) ? 16 : 0;
  int arow  = lane & 15;
  int ncol  = lane & 15;

  v8f acc[2][4];
#pragma unroll
  for (int i = 0; i < 2; ++i)
#pragma unroll
    for (int j = 0; j < 4; ++j) acc[i][j] = (v8f){0,0,0,0,0,0,0,0};

  const int r0 = (tid * 2)     >> 2, c0 = ((tid * 2)     & 3) << 3;
  const int r1 = (tid * 2 + 1) >> 2, c1 = ((tid * 2 + 1) & 3) << 3;
  const int hr0 = imin_(mb * 128 + r0, cnt - 1);
  const int hr1 = imin_(mb * 128 + r1, cnt - 1);
  const __bf16* ap0 = h + (size_t)hr0 * DFF + c0;
  const __bf16* ap1 = h + (size_t)hr1 * DFF + c1;
  const __bf16* bp0 = w2 + (size_t)(nb * 128 + r0) * DFF + c0;
  const __bf16* bp1 = w2 + (size_t)(nb * 128 + r1) * DFF + c1;

  {
    uint4 a0v = *(const uint4*)ap0;
    uint4 a1v = *(const uint4*)ap1;
    uint4 b0v = *(const uint4*)bp0;
    uint4 b1v = *(const uint4*)bp1;
    *(uint4*)(&As[0][r0][c0]) = a0v;
    *(uint4*)(&As[0][r1][c1]) = a1v;
    *(uint4*)(&Bs[0][r0][c0]) = b0v;
    *(uint4*)(&Bs[0][r1][c1]) = b1v;
  }
  __syncthreads();

  const int NK = DFF / 32;
  for (int ki = 0; ki < NK; ++ki){
    int cur = ki & 1, nxt = cur ^ 1;
    int kn = imin_((ki + 1) * 32, DFF - 32);
    uint4 a0v = *(const uint4*)(ap0 + kn);
    uint4 a1v = *(const uint4*)(ap1 + kn);
    uint4 b0v = *(const uint4*)(bp0 + kn);
    uint4 b1v = *(const uint4*)(bp1 + kn);
    *(uint4*)(&As[nxt][r0][c0]) = a0v;
    *(uint4*)(&As[nxt][r1][c1]) = a1v;
    *(uint4*)(&Bs[nxt][r0][c0]) = b0v;
    *(uint4*)(&Bs[nxt][r1][c1]) = b1v;

    v16bf a0 = frag_ld(&As[cur][mw * 32 + arow][aoff], 8);
    v16bf a1 = frag_ld(&As[cur][mw * 32 + 16 + arow][aoff], 8);
    v16bf bb0 = frag_ld(&Bs[cur][nw * 64 +  0 + ncol][koffB], 8);
    v16bf bb1 = frag_ld(&Bs[cur][nw * 64 + 16 + ncol][koffB], 8);
    v16bf bb2 = frag_ld(&Bs[cur][nw * 64 + 32 + ncol][koffB], 8);
    v16bf bb3 = frag_ld(&Bs[cur][nw * 64 + 48 + ncol][koffB], 8);
    acc[0][0] = wmma_bf16(a0, bb0, acc[0][0]);
    acc[0][1] = wmma_bf16(a0, bb1, acc[0][1]);
    acc[0][2] = wmma_bf16(a0, bb2, acc[0][2]);
    acc[0][3] = wmma_bf16(a0, bb3, acc[0][3]);
    acc[1][0] = wmma_bf16(a1, bb0, acc[1][0]);
    acc[1][1] = wmma_bf16(a1, bb1, acc[1][1]);
    acc[1][2] = wmma_bf16(a1, bb2, acc[1][2]);
    acc[1][3] = wmma_bf16(a1, bb3, acc[1][3]);
    __syncthreads();
  }

#pragma unroll
  for (int mi = 0; mi < 2; ++mi)
#pragma unroll
    for (int ni = 0; ni < 4; ++ni){
      int col = nb * 128 + nw * 64 + ni * 16 + (lane & 15);
      float bias = b2[col];
#pragma unroll
      for (int v = 0; v < 8; ++v){
        int row = mw * 32 + mi * 16 + v + ((lane & 16) ? 8 : 0);
        int gm = mb * 128 + row;
        if (gm < cnt){
          float val = sgate[row] * (acc[mi][ni][v] + bias);
          size_t oi = (size_t)stok[row] * DM + col;
          if (expert >= 0) val += out[oi];
          out[oi] = val;
        }
      }
    }
}

extern "C" void kernel_launch(void* const* d_in, const int* in_sizes, int n_in,
                              void* d_out, int out_size, void* d_ws, size_t ws_size,
                              hipStream_t stream){
  (void)in_sizes; (void)n_in; (void)out_size; (void)ws_size;
  const float* x   = (const float*)d_in[0];
  const float* sw1 = (const float*)d_in[1];
  const float* sb1 = (const float*)d_in[2];
  const float* sw2 = (const float*)d_in[3];
  const float* sb2 = (const float*)d_in[4];
  const float* rw  = (const float*)d_in[5];
  const float* rb  = (const float*)d_in[6];
  const float* ew1 = (const float*)d_in[7];
  const float* eb1 = (const float*)d_in[8];
  const float* ew2 = (const float*)d_in[9];
  const float* eb2 = (const float*)d_in[10];
  float* out = (float*)d_out;

  char* p = (char*)d_ws;
  size_t off = 0;
#define WSALLOC(ty, name, count) ty* name = (ty*)(p + off); off = (off + (size_t)(count) * sizeof(ty) + 255) & ~(size_t)255;
  WSALLOC(__bf16, xb,   (size_t)U_TOK * DM)
  WSALLOC(__bf16, wb1,  (size_t)9 * DFF * DM)
  WSALLOC(__bf16, wb2,  (size_t)9 * DM * DFF)
  WSALLOC(__bf16, hbuf, (size_t)U_TOK * DFF)
  WSALLOC(int,    tk_idx, U_TOK * 2)
  WSALLOC(float,  tk_w,   U_TOK * 2)
  WSALLOC(int,    etok,   U_TOK * 2)
  WSALLOC(float,  egate,  U_TOK * 2)
  WSALLOC(int,    eoff,   NEXP + 1)
#undef WSALLOC

  // fp32 -> bf16 conversions (x, expert weights, shared weights at slot 8)
  k_cvt<<<2048, 256, 0, stream>>>((const float4*)x,   (bf16x4*)xb,  U_TOK * DM / 4);
  k_cvt<<<4096, 256, 0, stream>>>((const float4*)ew1, (bf16x4*)wb1, NEXP * DFF * DM / 4);
  k_cvt<<<1024, 256, 0, stream>>>((const float4*)sw1, (bf16x4*)(wb1 + (size_t)NEXP * DFF * DM), DFF * DM / 4);
  k_cvt<<<4096, 256, 0, stream>>>((const float4*)ew2, (bf16x4*)wb2, NEXP * DM * DFF / 4);
  k_cvt<<<1024, 256, 0, stream>>>((const float4*)sw2, (bf16x4*)(wb2 + (size_t)NEXP * DM * DFF), DM * DFF / 4);

  // routing
  k_router<<<U_TOK / 8, 256, 0, stream>>>(x, rw, rb, tk_idx, tk_w);
  k_build<<<1, 256, 0, stream>>>(tk_idx, tk_w, etok, egate, eoff);

  // shared FFN (dense, initializes out)
  k_ffn1<<<dim3(U_TOK / 128, DFF / 128), 256, 0, stream>>>(
      xb, wb1 + (size_t)NEXP * DFF * DM, sb1, hbuf, etok, eoff, -1);
  k_ffn2<<<dim3(U_TOK / 128, DM / 128), 256, 0, stream>>>(
      hbuf, wb2 + (size_t)NEXP * DM * DFF, sb2, out, etok, egate, eoff, -1);

  // specialists: gathered grouped GEMMs, sequential accumulate (race-free)
  for (int e = 0; e < NEXP; ++e){
    k_ffn1<<<dim3(U_TOK / 128, DFF / 128), 256, 0, stream>>>(
        xb, wb1 + (size_t)e * DFF * DM, eb1 + (size_t)e * DFF, hbuf, etok, eoff, e);
    k_ffn2<<<dim3(U_TOK / 128, DM / 128), 256, 0, stream>>>(
        hbuf, wb2 + (size_t)e * DM * DFF, eb2 + (size_t)e * DM, out, etok, egate, eoff, e);
  }
}
